// HierarchicalSelfAttention_3410204033410
// MI455X (gfx1250) — matrix operand order repaired and checked by reading, not compiled, not run on hardware
//
#include <hip/hip_runtime.h>
#include <hip/hip_bf16.h>

typedef __attribute__((ext_vector_type(16))) _Float16 v16h;
typedef __attribute__((ext_vector_type(8)))  _Float16 v8h;
typedef __attribute__((ext_vector_type(8)))  float    v8f;

#define B_   2
#define L_   2048
#define E_   512
#define H_   8
#define HD_  64
#define E3_  1536
#define M_   (B_*L_)   /* 4096 */
#define WIN_ 128

__device__ __forceinline__ v16h join16(v8h lo, v8h hi) {
  v16h r;
#pragma unroll
  for (int i = 0; i < 8; ++i) { r[i] = lo[i]; r[i + 8] = hi[i]; }
  return r;
}

__device__ __forceinline__ v8h ldg8(const _Float16* p) { return *(const v8h*)p; }

// ---------------- kernel 0: f32 -> f16 convert ----------------
__global__ void k_convert(const float* __restrict__ src, _Float16* __restrict__ dst, int n) {
  int i = blockIdx.x * blockDim.x + threadIdx.x;
  int stride = gridDim.x * blockDim.x;
  for (; i < n; i += stride) dst[i] = (_Float16)src[i];
}

// ---------------- kernel 1: QKV projection (both modules) ----------------
// qkv[m][n] = sum_k x[m][k] * w_in[n][k] + b_in[n]; also store V transposed as [b,h,d,L].
__global__ __launch_bounds__(128)
void k_qkv(const _Float16* __restrict__ xh,
           const _Float16* __restrict__ win0, const _Float16* __restrict__ win1,
           const float* __restrict__ bin0, const float* __restrict__ bin1,
           _Float16* __restrict__ qkv0, _Float16* __restrict__ qkv1,
           _Float16* __restrict__ vT0,  _Float16* __restrict__ vT1) {
  const int lane = threadIdx.x & 31;
  const int wv   = threadIdx.x >> 5;
  const int l15  = lane & 15;
  const int half = lane >> 4;

  int t = blockIdx.x * 4 + wv;            // 0 .. 2*256*96-1
  const int mod = t / (256 * 96);
  t -= mod * 256 * 96;
  const int rowTile = t / 96;
  const int colTile = t - rowTile * 96;
  const int m0 = rowTile * 16;
  const int n0 = colTile * 16;

  const _Float16* w   = mod ? win1 : win0;
  const float*    bin = mod ? bin1 : bin0;
  _Float16* qkv = mod ? qkv1 : qkv0;
  _Float16* vT  = mod ? vT1  : vT0;

  const _Float16* arow = xh + (m0 + l15) * E_;
  const _Float16* brow = w  + (n0 + l15) * E_;

  v8f c = {};
#pragma unroll 4
  for (int kk = 0; kk < E_; kk += 32) {
    v16h a  = join16(ldg8(arow + kk + 16 * half), ldg8(arow + kk + 16 * half + 8));
    v16h bm = join16(ldg8(brow + kk + 16 * half), ldg8(brow + kk + 16 * half + 8));
    c = __builtin_amdgcn_wmma_f32_16x16x32_f16(false, a, false, bm, (short)0, c, false, false);
  }
  const int   nCol = n0 + l15;
  const float bias = bin[nCol];
#pragma unroll
  for (int r = 0; r < 8; ++r) {
    const int m = m0 + r + 8 * half;
    const float v = c[r] + bias;
    qkv[(size_t)m * E3_ + nCol] = (_Float16)v;
    if (nCol >= 2 * E_) {                 // V part -> transposed copy
      const int nv = nCol - 2 * E_;
      const int h = nv >> 6, d = nv & 63;
      const int b = m >> 11, li = m & (L_ - 1);
      vT[((size_t)(b * H_ + h) * HD_ + d) * L_ + li] = (_Float16)v;
    }
  }
}

// ---------------- kernel 2: flash attention (local + global) ----------------
__global__ __launch_bounds__(128)
void k_attn(const _Float16* __restrict__ qkv0, const _Float16* __restrict__ qkv1,
            const _Float16* __restrict__ vT0,  const _Float16* __restrict__ vT1,
            const unsigned char* __restrict__ kpad,
            _Float16* __restrict__ att0, _Float16* __restrict__ att1) {
  __shared__ __align__(16) _Float16 Plds[4][16][32];

  const int lane = threadIdx.x & 31;
  const int wv   = threadIdx.x >> 5;
  const int l15  = lane & 15;
  const int half = lane >> 4;

  const int x   = blockIdx.x;
  const int qg  = x & 31;
  const int h   = (x >> 5) & 7;
  const int b   = (x >> 8) & 1;
  const int mod = x >> 9;
  const bool isLocal = (mod == 0);

  const _Float16* qkv = mod ? qkv1 : qkv0;
  const _Float16* vTt = mod ? vT1  : vT0;
  _Float16*       att = mod ? att1 : att0;

  const int m0 = qg * 64 + wv * 16;
  const int g0 = b * L_ + m0;

  const _Float16* qrow = qkv + (size_t)(g0 + l15) * E3_ + h * HD_;
  const v16h aQ0 = join16(ldg8(qrow + 16 * half),      ldg8(qrow + 16 * half + 8));
  const v16h aQ1 = join16(ldg8(qrow + 32 + 16 * half), ldg8(qrow + 32 + 16 * half + 8));

  float row_max[8], row_sum[8];
#pragma unroll
  for (int r = 0; r < 8; ++r) { row_max[r] = -3.0e38f; row_sum[r] = 0.0f; }
  v8f o0 = {}, o1 = {}, o2 = {}, o3 = {};

  int cstart = 0, cend = L_;
  if (isLocal) {
    cstart = (m0 > WIN_) ? ((m0 - WIN_) & ~31) : 0;
    int ce = (m0 + 16 + WIN_ + 31) & ~31;
    cend = ce < L_ ? ce : L_;
  }

  const _Float16* vbase = vTt + (size_t)(b * H_ + h) * HD_ * L_;

  for (int c0 = cstart; c0 < cend; c0 += 32) {
    // ---- scores S = Q K^T for 32 keys (two 16-key tiles) ----
    const _Float16* k0row = qkv + (size_t)(b * L_ + c0 +      l15) * E3_ + E_ + h * HD_;
    const _Float16* k1row = qkv + (size_t)(b * L_ + c0 + 16 + l15) * E3_ + E_ + h * HD_;
    const v16h kb00 = join16(ldg8(k0row + 16 * half),      ldg8(k0row + 16 * half + 8));
    const v16h kb01 = join16(ldg8(k0row + 32 + 16 * half), ldg8(k0row + 32 + 16 * half + 8));
    const v16h kb10 = join16(ldg8(k1row + 16 * half),      ldg8(k1row + 16 * half + 8));
    const v16h kb11 = join16(ldg8(k1row + 32 + 16 * half), ldg8(k1row + 32 + 16 * half + 8));
    v8f s0 = {}, s1 = {};
    s0 = __builtin_amdgcn_wmma_f32_16x16x32_f16(false, aQ0, false, kb00, (short)0, s0, false, false);
    s0 = __builtin_amdgcn_wmma_f32_16x16x32_f16(false, aQ1, false, kb01, (short)0, s0, false, false);
    s1 = __builtin_amdgcn_wmma_f32_16x16x32_f16(false, aQ0, false, kb10, (short)0, s1, false, false);
    s1 = __builtin_amdgcn_wmma_f32_16x16x32_f16(false, aQ1, false, kb11, (short)0, s1, false, false);

    const int key0 = c0 + l15;
    const int key1 = c0 + 16 + l15;
    const bool kp0 = kpad[b * L_ + key0] != 0;
    const bool kp1 = kpad[b * L_ + key1] != 0;

    // ---- mask + online softmax (row stats per C-fragment row) ----
#pragma unroll
    for (int r = 0; r < 8; ++r) {
      float v0 = s0[r] * 0.125f;   // 1/sqrt(64)
      float v1 = s1[r] * 0.125f;
      if (isLocal) {
        const int m = m0 + r + 8 * half;
        int d0 = m - key0; d0 = d0 < 0 ? -d0 : d0;
        int d1 = m - key1; d1 = d1 < 0 ? -d1 : d1;
        if (d0 > WIN_) v0 = -3.0e38f;
        if (d1 > WIN_) v1 = -3.0e38f;
      }
      if (kp0) v0 = -3.0e38f;
      if (kp1) v1 = -3.0e38f;

      float mx = fmaxf(v0, v1);
#pragma unroll
      for (int sh = 1; sh < 16; sh <<= 1) mx = fmaxf(mx, __shfl_xor(mx, sh, 32));
      const float nm = fmaxf(row_max[r], mx);
      const float cr = __expf(row_max[r] - nm);
      row_max[r] = nm;
      const float p0 = __expf(v0 - nm);
      const float p1 = __expf(v1 - nm);
      float ps = p0 + p1;
#pragma unroll
      for (int sh = 1; sh < 16; sh <<= 1) ps += __shfl_xor(ps, sh, 32);
      row_sum[r] = row_sum[r] * cr + ps;
      o0[r] *= cr; o1[r] *= cr; o2[r] *= cr; o3[r] *= cr;
      const int mr = r + 8 * half;
      Plds[wv][mr][l15]      = (_Float16)p0;   // C-layout -> LDS
      Plds[wv][mr][16 + l15] = (_Float16)p1;
    }

    // re-read P in A-fragment layout (per-wave LDS slice; DS ops are in-order per wave)
    const _Float16* prow = &Plds[wv][l15][0];
    const v16h aP = join16(*(const v8h*)(prow + 16 * half), *(const v8h*)(prow + 16 * half + 8));

    // ---- O += P V, V read from transposed layout (contiguous keys) ----
    const _Float16* v0p = vbase + (size_t)(0 * 16 + l15) * L_ + c0 + 16 * half;
    const _Float16* v1p = vbase + (size_t)(1 * 16 + l15) * L_ + c0 + 16 * half;
    const _Float16* v2p = vbase + (size_t)(2 * 16 + l15) * L_ + c0 + 16 * half;
    const _Float16* v3p = vbase + (size_t)(3 * 16 + l15) * L_ + c0 + 16 * half;
    o0 = __builtin_amdgcn_wmma_f32_16x16x32_f16(false, aP, false, join16(ldg8(v0p), ldg8(v0p + 8)), (short)0, o0, false, false);
    o1 = __builtin_amdgcn_wmma_f32_16x16x32_f16(false, aP, false, join16(ldg8(v1p), ldg8(v1p + 8)), (short)0, o1, false, false);
    o2 = __builtin_amdgcn_wmma_f32_16x16x32_f16(false, aP, false, join16(ldg8(v2p), ldg8(v2p + 8)), (short)0, o2, false, false);
    o3 = __builtin_amdgcn_wmma_f32_16x16x32_f16(false, aP, false, join16(ldg8(v3p), ldg8(v3p + 8)), (short)0, o3, false, false);
  }

#pragma unroll
  for (int r = 0; r < 8; ++r) {
    const float inv = 1.0f / row_sum[r];
    _Float16* orow = att + (size_t)(g0 + r + 8 * half) * E_ + h * HD_;
    orow[0 * 16 + l15] = (_Float16)(o0[r] * inv);
    orow[1 * 16 + l15] = (_Float16)(o1[r] * inv);
    orow[2 * 16 + l15] = (_Float16)(o2[r] * inv);
    orow[3 * 16 + l15] = (_Float16)(o3[r] * inv);
  }
}

// ---------------- kernel 3: fused output projections + gate ----------------
__global__ __launch_bounds__(128)
void k_outproj(const _Float16* __restrict__ attL, const _Float16* __restrict__ attG,
               const _Float16* __restrict__ wL, const _Float16* __restrict__ wG,
               const float* __restrict__ bL, const float* __restrict__ bG,
               const float* __restrict__ gate_p,
               float* __restrict__ out) {
  const int lane = threadIdx.x & 31;
  const int wv   = threadIdx.x >> 5;
  const int l15  = lane & 15;
  const int half = lane >> 4;

  const int t = blockIdx.x * 4 + wv;     // 0..8191
  const int m0 = (t >> 5) * 16;
  const int n0 = (t & 31) * 16;

  const _Float16* aLrow = attL + (size_t)(m0 + l15) * E_;
  const _Float16* aGrow = attG + (size_t)(m0 + l15) * E_;
  const _Float16* bLrow = wL + (size_t)(n0 + l15) * E_;
  const _Float16* bGrow = wG + (size_t)(n0 + l15) * E_;

  v8f cl = {}, cg = {};
#pragma unroll 4
  for (int kk = 0; kk < E_; kk += 32) {
    const v16h aL  = join16(ldg8(aLrow + kk + 16 * half), ldg8(aLrow + kk + 16 * half + 8));
    const v16h bLm = join16(ldg8(bLrow + kk + 16 * half), ldg8(bLrow + kk + 16 * half + 8));
    cl = __builtin_amdgcn_wmma_f32_16x16x32_f16(false, aL, false, bLm, (short)0, cl, false, false);
    const v16h aG  = join16(ldg8(aGrow + kk + 16 * half), ldg8(aGrow + kk + 16 * half + 8));
    const v16h bGm = join16(ldg8(bGrow + kk + 16 * half), ldg8(bGrow + kk + 16 * half + 8));
    cg = __builtin_amdgcn_wmma_f32_16x16x32_f16(false, aG, false, bGm, (short)0, cg, false, false);
  }
  const float g = gate_p[0];
  const float biasL = bL[n0 + l15];
  const float biasG = bG[n0 + l15];
#pragma unroll
  for (int r = 0; r < 8; ++r) {
    const int m = m0 + r + 8 * half;
    out[(size_t)m * E_ + n0 + l15] = g * (cl[r] + biasL) + (1.0f - g) * (cg[r] + biasG);
  }
}

extern "C" void kernel_launch(void* const* d_in, const int* in_sizes, int n_in,
                              void* d_out, int out_size, void* d_ws, size_t ws_size,
                              hipStream_t stream) {
  (void)in_sizes; (void)n_in; (void)out_size; (void)ws_size;
  const float* x     = (const float*)d_in[0];
  const unsigned char* kpad = (const unsigned char*)d_in[1];
  const float* lwin  = (const float*)d_in[2];
  const float* lbin  = (const float*)d_in[3];
  const float* lwout = (const float*)d_in[4];
  const float* lbout = (const float*)d_in[5];
  const float* gwin  = (const float*)d_in[6];
  const float* gbin  = (const float*)d_in[7];
  const float* gwout = (const float*)d_in[8];
  const float* gbout = (const float*)d_in[9];
  const float* gate  = (const float*)d_in[10];
  float* out = (float*)d_out;

  char* ws = (char*)d_ws;
  size_t off = 0;
  auto carve = [&](size_t nhalf) {
    _Float16* p = (_Float16*)(ws + off);
    off += nhalf * sizeof(_Float16);
    return p;
  };
  _Float16* xh     = carve((size_t)M_ * E_);
  _Float16* winh0  = carve((size_t)E3_ * E_);
  _Float16* winh1  = carve((size_t)E3_ * E_);
  _Float16* wouth0 = carve((size_t)E_ * E_);
  _Float16* wouth1 = carve((size_t)E_ * E_);
  _Float16* qkv0   = carve((size_t)M_ * E3_);
  _Float16* qkv1   = carve((size_t)M_ * E3_);
  _Float16* vT0    = carve((size_t)B_ * H_ * HD_ * L_);
  _Float16* vT1    = carve((size_t)B_ * H_ * HD_ * L_);
  _Float16* att0   = carve((size_t)M_ * E_);
  _Float16* att1   = carve((size_t)M_ * E_);

  k_convert<<<1024, 256, 0, stream>>>(x, xh, M_ * E_);
  k_convert<<<512, 256, 0, stream>>>(lwin, winh0, E3_ * E_);
  k_convert<<<512, 256, 0, stream>>>(gwin, winh1, E3_ * E_);
  k_convert<<<256, 256, 0, stream>>>(lwout, wouth0, E_ * E_);
  k_convert<<<256, 256, 0, stream>>>(gwout, wouth1, E_ * E_);

  k_qkv<<<12288, 128, 0, stream>>>(xh, winh0, winh1, lbin, gbin, qkv0, qkv1, vT0, vT1);
  k_attn<<<1024, 128, 0, stream>>>(qkv0, qkv1, vT0, vT1, kpad, att0, att1);
  k_outproj<<<2048, 128, 0, stream>>>(att0, att1, wouth0, wouth1, lbout, gbout, gate, out);
}
